// TransformerNet_42460046688745
// MI455X (gfx1250) — matrix operand order repaired and checked by reading, not compiled, not run on hardware
//
#include <hip/hip_runtime.h>

#define DEV static __device__ __forceinline__

typedef __attribute__((ext_vector_type(16))) __bf16 v16bf;
typedef __attribute__((ext_vector_type(8)))  __bf16 v8bf;
typedef __attribute__((ext_vector_type(8)))  float  v8f;

constexpr int Bn = 4, Tn = 2048, Dn = 128, Hn = 4, DKn = 32, NBn = 2;
constexpr int Mn = Bn * Tn;                         // 8192 token rows
constexpr float LN_EPS = 1e-3f;
// Q is pre-scaled by 1/sqrt(DK) * log2(e): softmax runs in base-2 domain.
constexpr float Q_PRESCALE = 0.25503486f;

// ---------------------------------------------------------------- helpers
DEV __bf16 f2bf(float f) {                          // round-to-nearest-even
  unsigned u = __builtin_bit_cast(unsigned, f);
  u += 0x7FFFu + ((u >> 16) & 1u);
  unsigned short h = (unsigned short)(u >> 16);
  return __builtin_bit_cast(__bf16, h);
}

DEV unsigned pack2(float lo, float hi) {            // two bf16 in a dword
  unsigned l = __builtin_bit_cast(unsigned short, f2bf(lo));
  unsigned hh = __builtin_bit_cast(unsigned short, f2bf(hi));
  return (hh << 16) | l;
}

DEV v8bf ld8(const __bf16* p) { return *(const v8bf*)p; }

DEV v16bf join8(v8bf lo, v8bf hi) {
  return __builtin_shufflevector(lo, hi, 0, 1, 2, 3, 4, 5, 6, 7,
                                 8, 9, 10, 11, 12, 13, 14, 15);
}

DEV v8f wmma_bf16(v16bf a, v16bf b, v8f c) {
  return __builtin_amdgcn_wmma_f32_16x16x32_bf16(false, a, false, b,
                                                 (short)0, c, false, false);
}

// CDNA5 async global->LDS copy (ASYNCcnt); per-lane 16B.
DEV void async_load_b128(unsigned lds_addr, unsigned long long gaddr) {
  asm volatile("global_load_async_to_lds_b128 %0, %1, off"
               :: "v"(lds_addr), "v"(gaddr) : "memory");
}
#define WAIT_ASYNC0() asm volatile("s_wait_asynccnt 0x0" ::: "memory")

DEV unsigned lds_off(const void* p) {               // ISA 10.2: low 32 bits
  return (unsigned)(uintptr_t)p;
}

// ---------------------------------------------------------------- posenc
__global__ __launch_bounds__(256) void posenc_kernel(const float* __restrict__ in,
                                                     float* __restrict__ out) {
  int idx = blockIdx.x * 256 + threadIdx.x;         // < Mn*Dn
  int i = idx & (Dn - 1);
  int t = (idx >> 7) & (Tn - 1);
  float arg = (float)t * __expf(-0.14391156831f * (float)i); // ln(1e4)/64
  float pe  = (i & 1) ? __cosf(arg) : __sinf(arg);
  out[idx] = in[idx] + pe;
}

// ---------------------------------------------------------------- weights
// f32 [nb,K,N] -> bf16 transposed [nb,N,K] so B-fragments load contiguously
__global__ __launch_bounds__(256) void cvtT_kernel(const float* __restrict__ x,
                                                   __bf16* __restrict__ y, int n) {
  int i = blockIdx.x * 256 + threadIdx.x;
  if (i < n) {
    int nb = i >> 14;
    int r = (i >> 7) & (Dn - 1);                    // output row  = N
    int c = i & (Dn - 1);                           // output col  = K
    y[i] = f2bf(x[(nb << 14) + c * Dn + r]);
  }
}

// ---------------------------------------------------------------- LayerNorm
template <bool WB>
__global__ __launch_bounds__(256) void ln_kernel(const float* __restrict__ x,
                                                 const float* __restrict__ g,
                                                 const float* __restrict__ bb,
                                                 float* __restrict__ yf,
                                                 __bf16* __restrict__ yb) {
  int row  = blockIdx.x * 8 + (threadIdx.x >> 5);
  int lane = threadIdx.x & 31;
  float4 v = ((const float4*)(x + (size_t)row * Dn))[lane];
  float s = v.x + v.y + v.z + v.w;
#pragma unroll
  for (int m = 16; m >= 1; m >>= 1) s += __shfl_xor(s, m, 32);
  float mu = s * (1.0f / Dn);
  float dx = v.x - mu, dy = v.y - mu, dz = v.z - mu, dw = v.w - mu;
  float q = dx * dx + dy * dy + dz * dz + dw * dw;
#pragma unroll
  for (int m = 16; m >= 1; m >>= 1) q += __shfl_xor(q, m, 32);
  float rstd = rsqrtf(q * (1.0f / Dn) + LN_EPS);
  float4 gg = ((const float4*)g)[lane];
  float4 bv = ((const float4*)bb)[lane];
  float4 y;
  y.x = gg.x * dx * rstd + bv.x;
  y.y = gg.y * dy * rstd + bv.y;
  y.z = gg.z * dz * rstd + bv.z;
  y.w = gg.w * dw * rstd + bv.w;
  ((float4*)(yf + (size_t)row * Dn))[lane] = y;
  if constexpr (WB) {
    uint2 pk;
    pk.x = pack2(y.x, y.y);
    pk.y = pack2(y.z, y.w);
    ((uint2*)(yb + (size_t)row * Dn))[lane] = pk;
  }
}

// ---------------------------------------------------------------- epilogues
DEV void scatter_bhtd(__bf16* __restrict__ out, const v8f& acc, float bv,
                      float scl, int gm0, int gn, int sel) {
  int h_ = gn >> 5, d_ = gn & (DKn - 1);
#pragma unroll
  for (int r = 0; r < 8; ++r) {
    int gm = gm0 + r + sel * 8;
    int b_ = gm >> 11, t_ = gm & (Tn - 1);
    out[(((size_t)(b_ * Hn + h_) * Tn) + t_) * DKn + d_] =
        f2bf((acc[r] + bv) * scl);
  }
}

DEV void store_vt(__bf16* __restrict__ out, const v8f& acc, float bv,
                  int gm0, int gn, int sel) {
  int b_ = gm0 >> 11;                               // tile never spans batches
  int t0 = (gm0 + sel * 8) & (Tn - 1);
  int hh = gn >> 5, dk = gn & (DKn - 1);
  uint4 w;
  w.x = pack2(acc[0] + bv, acc[1] + bv);
  w.y = pack2(acc[2] + bv, acc[3] + bv);
  w.z = pack2(acc[4] + bv, acc[5] + bv);
  w.w = pack2(acc[6] + bv, acc[7] + bv);
  *(uint4*)(out + ((size_t)((b_ * Hn + hh) * DKn) + dk) * Tn + t0) = w;
}

// ---------------------------------------------------------------- fused QKV
// One pass over A feeds 3 accumulator sets (24 WMMAs/wave, 2 M-tiles).
// Q is pre-scaled by Q_PRESCALE; V is stored transposed [B,H,DK,T].
__global__ __launch_bounds__(256) void qkv_kernel(const __bf16* __restrict__ A,
                                                  const __bf16* __restrict__ WqT,
                                                  const __bf16* __restrict__ WkT,
                                                  const __bf16* __restrict__ WvT,
                                                  const float* __restrict__ bq,
                                                  const float* __restrict__ bk,
                                                  const float* __restrict__ bv,
                                                  __bf16* __restrict__ Qo,
                                                  __bf16* __restrict__ Ko,
                                                  __bf16* __restrict__ Vto) {
  const int lane = threadIdx.x & 31;
  const int wid  = threadIdx.x >> 5;
  const int n0   = wid * 16;
  const int l15  = lane & 15;
  const int sel  = lane >> 4;
  const int m0   = blockIdx.x * 32;

  const __bf16* ar0 = A + (size_t)(m0 + l15) * Dn;
  const __bf16* ar1 = A + (size_t)(m0 + 16 + l15) * Dn;
  const __bf16* wq  = WqT + (size_t)(n0 + l15) * Dn;
  const __bf16* wk  = WkT + (size_t)(n0 + l15) * Dn;
  const __bf16* wv  = WvT + (size_t)(n0 + l15) * Dn;

  v8f q0a = (v8f)0.0f, q1a = (v8f)0.0f;
  v8f k0a = (v8f)0.0f, k1a = (v8f)0.0f;
  v8f v0a = (v8f)0.0f, v1a = (v8f)0.0f;
#pragma unroll
  for (int kk = 0; kk < Dn; kk += 32) {
    v16bf a0 = join8(ld8(ar0 + kk + sel * 16), ld8(ar0 + kk + sel * 16 + 8));
    v16bf a1 = join8(ld8(ar1 + kk + sel * 16), ld8(ar1 + kk + sel * 16 + 8));
    v16bf fq = join8(ld8(wq + kk + sel * 16), ld8(wq + kk + sel * 16 + 8));
    v16bf fk = join8(ld8(wk + kk + sel * 16), ld8(wk + kk + sel * 16 + 8));
    v16bf fv = join8(ld8(wv + kk + sel * 16), ld8(wv + kk + sel * 16 + 8));
    q0a = wmma_bf16(a0, fq, q0a);
    q1a = wmma_bf16(a1, fq, q1a);
    k0a = wmma_bf16(a0, fk, k0a);
    k1a = wmma_bf16(a1, fk, k1a);
    v0a = wmma_bf16(a0, fv, v0a);
    v1a = wmma_bf16(a1, fv, v1a);
  }
  const int gn = n0 + l15;
  scatter_bhtd(Qo, q0a, bq[gn], Q_PRESCALE, m0, gn, sel);
  scatter_bhtd(Qo, q1a, bq[gn], Q_PRESCALE, m0 + 16, gn, sel);
  scatter_bhtd(Ko, k0a, bk[gn], 1.0f, m0, gn, sel);
  scatter_bhtd(Ko, k1a, bk[gn], 1.0f, m0 + 16, gn, sel);
  store_vt(Vto, v0a, bv[gn], m0, gn, sel);
  store_vt(Vto, v1a, bv[gn], m0 + 16, gn, sel);
}

// ---------------------------------------------------------------- GEMM
// C[M,128] = A @ W + bias; W TRANSPOSED bf16 [128N,128K]; 2 M-tiles/wave.
// MODE 0: f32 store   1: bf16 + ReLU   3: f32 + residual
template <int MODE>
__global__ __launch_bounds__(256) void gemm128_kernel(const __bf16* __restrict__ A,
                                                      const __bf16* __restrict__ Wt,
                                                      const float*  __restrict__ bias,
                                                      void*         __restrict__ outp,
                                                      const float*  __restrict__ res) {
  const int lane = threadIdx.x & 31;
  const int wid  = threadIdx.x >> 5;
  const int n0   = wid * 16;
  const int l15  = lane & 15;
  const int sel  = lane >> 4;
  const int m0   = blockIdx.x * 32;

  const __bf16* ar0 = A + (size_t)(m0 + l15) * Dn;
  const __bf16* ar1 = A + (size_t)(m0 + 16 + l15) * Dn;
  const __bf16* wr  = Wt + (size_t)(n0 + l15) * Dn;

  v8f acc0 = (v8f)0.0f, acc1 = (v8f)0.0f;
#pragma unroll
  for (int kk = 0; kk < Dn; kk += 32) {
    v16bf a0 = join8(ld8(ar0 + kk + sel * 8), ld8(ar0 + kk + 16 + sel * 8));
    v16bf a1 = join8(ld8(ar1 + kk + sel * 8), ld8(ar1 + kk + 16 + sel * 8));
    v16bf b  = join8(ld8(wr + kk + sel * 8), ld8(wr + kk + 16 + sel * 8));
    acc0 = wmma_bf16(a0, b, acc0);
    acc1 = wmma_bf16(a1, b, acc1);
  }
  const int gn = n0 + l15;
  const float bv = bias[gn];
#pragma unroll
  for (int t = 0; t < 2; ++t) {
    const v8f& acc = t ? acc1 : acc0;
    const int gmb = m0 + t * 16;
#pragma unroll
    for (int r = 0; r < 8; ++r) {
      const int gm = gmb + r + sel * 8;
      float v = acc[r] + bv;
      if constexpr (MODE == 0) {
        ((float*)outp)[(size_t)gm * Dn + gn] = v;
      } else if constexpr (MODE == 1) {
        ((__bf16*)outp)[(size_t)gm * Dn + gn] = f2bf(fmaxf(v, 0.0f));
      } else {
        ((float*)outp)[(size_t)gm * Dn + gn] = v + res[(size_t)gm * Dn + gn];
      }
    }
  }
}

// ---------------------------------------------------------------- attention
// Flash-style; one 16-query tile per wave; all 8 waves of a block share one
// (b,h), so K/V 32-key chunks are staged once per block into LDS with the
// CDNA5 async-to-LDS path, double buffered (DMA overlaps WMMA+softmax).
// Q pre-scaled by 1/sqrt(DK)*log2(e): softmax in base-2 (exp2), same values.
__global__ __launch_bounds__(256) void attn_kernel(const __bf16* __restrict__ Q,
                                                   const __bf16* __restrict__ K,
                                                   const __bf16* __restrict__ Vt,
                                                   __bf16* __restrict__ Out) {
  __shared__ __align__(16) __bf16 sK[2][32 * 32];   // [key][dk]   2 KB each
  __shared__ __align__(16) __bf16 sV[2][32 * 32];   // [dk][key]   2 KB each

  const int tid  = threadIdx.x;
  const int lane = tid & 31;
  const int wid  = tid >> 5;
  const int gw   = blockIdx.x * 8 + wid;   // 0 .. B*H*T/16 - 1
  const int qt   = gw & (Tn / 16 - 1);
  const int bh   = gw >> 7;                // same for all 8 waves of a block
  const int q0   = qt * 16;
  const int l15  = lane & 15;
  const int sel  = lane >> 4;
  const int base = bh * Tn;

  // per-thread 16B staging assignment (waves 0-3: K, waves 4-7: V)
  const bool isK = tid < 128;
  const int  j   = tid & 127;
  const int  dkr = j >> 2, seg = j & 3;
  const unsigned long long gK =
      (unsigned long long)(K + (size_t)base * DKn) + (size_t)tid * 16;
  const unsigned long long gV =
      (unsigned long long)(Vt + (size_t)(bh * DKn + dkr) * Tn) + (size_t)seg * 16;
  const unsigned lK = lds_off((const char*)sK[0] + (size_t)tid * 16);
  const unsigned lV = lds_off((const char*)sV[0] + (size_t)dkr * 64 + seg * 16);

  auto issue = [&](int kc, int p) {       // uniform per wave
    if (isK) async_load_b128(lK + p * 2048, gK + (size_t)kc * 64);
    else     async_load_b128(lV + p * 2048, gV + (size_t)kc * 2);
  };
  issue(0, 0);

  // Q^T as B fragment: dk 8*sel..+7 and 16+8*sel..+7 per lane
  const __bf16* qrow = Q + (size_t)(base + q0 + l15) * DKn + sel * 8;
  v16bf bq = join8(ld8(qrow), ld8(qrow + 16));

  float rm = -1e30f, rl = 0.0f;            // base-2 running max / linear sum
  v8f acc0 = (v8f)0.0f, acc1 = (v8f)0.0f;  // ctx^T, dk 0-15 / 16-31

  for (int kc = 0; kc < Tn; kc += 32) {
    const int p = (kc >> 5) & 1;
    WAIT_ASYNC0();                         // own chunk-kc loads complete
    __syncthreads();                       // whole block's loads complete
    if (kc + 32 < Tn) issue(kc + 32, p ^ 1);

    const __bf16* kb = sK[p];
    const __bf16* vb = sV[p];
    v16bf ak0 = join8(ld8(kb + l15 * 32 + sel * 8),
                      ld8(kb + l15 * 32 + 16 + sel * 8));
    v16bf ak1 = join8(ld8(kb + (16 + l15) * 32 + sel * 8),
                      ld8(kb + (16 + l15) * 32 + 16 + sel * 8));
    v8f s0 = wmma_bf16(ak0, bq, (v8f)0.0f);  // rows=keys, cols=queries (base2)
    v8f s1 = wmma_bf16(ak1, bq, (v8f)0.0f);

    float smax = -1e30f;
#pragma unroll
    for (int r = 0; r < 8; ++r) smax = fmaxf(smax, fmaxf(s0[r], s1[r]));
    smax = fmaxf(smax, __shfl_xor(smax, 16, 32));   // merge key halves
    float mn = fmaxf(rm, smax);
    float corr = exp2f(rm - mn);

    float ps = 0.0f;
    unsigned d0[4], d1[4];
#pragma unroll
    for (int jj = 0; jj < 4; ++jj) {
      float a0 = exp2f(s0[2 * jj] - mn), b0 = exp2f(s0[2 * jj + 1] - mn);
      float a1 = exp2f(s1[2 * jj] - mn), b1 = exp2f(s1[2 * jj + 1] - mn);
      ps += (a0 + b0) + (a1 + b1);
      d0[jj] = pack2(a0, b0);
      d1[jj] = pack2(a1, b1);
    }
    ps += __shfl_xor(ps, 16, 32);
    rl = rl * corr + ps;
    rm = mn;
#pragma unroll
    for (int r = 0; r < 8; ++r) {
      acc0[r] = acc0[r] * corr;
      acc1[r] = acc1[r] * corr;
    }
    // half-wave swap of packed exp pairs -> P^T B-fragment (8 shuffles)
    union { unsigned u[8]; v16bf v; } pb;
#pragma unroll
    for (int jj = 0; jj < 4; ++jj) {
      unsigned o0 = (unsigned)__shfl_xor((int)d0[jj], 16, 32);
      unsigned o1 = (unsigned)__shfl_xor((int)d1[jj], 16, 32);
      pb.u[jj]     = sel ? o1 : d0[jj];
      pb.u[jj + 4] = sel ? d1[jj] : o0;
    }
    v16bf av0 = join8(ld8(vb + l15 * 32 + sel * 16),
                      ld8(vb + l15 * 32 + sel * 16 + 8));
    v16bf av1 = join8(ld8(vb + (16 + l15) * 32 + sel * 16),
                      ld8(vb + (16 + l15) * 32 + sel * 16 + 8));
    acc0 = wmma_bf16(av0, pb.v, acc0);
    acc1 = wmma_bf16(av1, pb.v, acc1);
  }

  // softmax divide + the source's renormalize-by-(sum+1e-8) fold
  float inv = 1.0f / (rl * (1.0f + 1e-8f));
  int b_ = bh >> 2, h_ = bh & 3;
  int orow = b_ * Tn + q0 + l15;           // token row; ctx^T col = query
  uint4 w0, w1;
  w0.x = pack2(acc0[0] * inv, acc0[1] * inv);
  w0.y = pack2(acc0[2] * inv, acc0[3] * inv);
  w0.z = pack2(acc0[4] * inv, acc0[5] * inv);
  w0.w = pack2(acc0[6] * inv, acc0[7] * inv);
  w1.x = pack2(acc1[0] * inv, acc1[1] * inv);
  w1.y = pack2(acc1[2] * inv, acc1[3] * inv);
  w1.z = pack2(acc1[4] * inv, acc1[5] * inv);
  w1.w = pack2(acc1[6] * inv, acc1[7] * inv);
  *(uint4*)(Out + (size_t)orow * Dn + h_ * DKn + sel * 8) = w0;       // dk 0-15
  *(uint4*)(Out + (size_t)orow * Dn + h_ * DKn + 16 + sel * 8) = w1;  // dk 16-31
}

// ---------------------------------------------------------------- launcher
extern "C" void kernel_launch(void* const* d_in, const int* in_sizes, int n_in,
                              void* d_out, int out_size, void* d_ws, size_t ws_size,
                              hipStream_t stream) {
  (void)in_sizes; (void)n_in; (void)out_size; (void)ws_size;
  const float* inputs = (const float*)d_in[0];
  const float* W[6]   = {(const float*)d_in[1], (const float*)d_in[3],
                         (const float*)d_in[5], (const float*)d_in[7],
                         (const float*)d_in[9], (const float*)d_in[11]}; // Wq Wk Wv Wo W1 W2
  const float* bia[6] = {(const float*)d_in[2], (const float*)d_in[4],
                         (const float*)d_in[6], (const float*)d_in[8],
                         (const float*)d_in[10], (const float*)d_in[12]};
  const float* ln_g = (const float*)d_in[13];
  const float* ln_b = (const float*)d_in[14];

  char* p = (char*)d_ws;
  auto carve = [&](size_t bytes) {
    char* r = p;
    p += (bytes + 255) & ~(size_t)255;
    return (void*)r;
  };
  __bf16* wbf[6];
  for (int j = 0; j < 6; ++j) wbf[j] = (__bf16*)carve((size_t)NBn * Dn * Dn * 2);
  float*  h    = (float*) carve((size_t)Mn * Dn * 4);
  float*  lnf  = (float*) carve((size_t)Mn * Dn * 4);
  __bf16* lnb  = (__bf16*)carve((size_t)Mn * Dn * 2);
  __bf16* qb   = (__bf16*)carve((size_t)Mn * Dn * 2);  // [B,H,T,DK], pre-scaled
  __bf16* kb   = (__bf16*)carve((size_t)Mn * Dn * 2);  // [B,H,T,DK]
  __bf16* vtb  = (__bf16*)carve((size_t)Mn * Dn * 2);  // [B,H,DK,T]
  __bf16* ctx  = (__bf16*)carve((size_t)Mn * Dn * 2);  // [B,T,D]
  __bf16* ffn1 = (__bf16*)carve((size_t)Mn * Dn * 2);

  const int wN = NBn * Dn * Dn;
  for (int j = 0; j < 6; ++j)
    cvtT_kernel<<<(wN + 255) / 256, 256, 0, stream>>>(W[j], wbf[j], wN);

  posenc_kernel<<<Mn * Dn / 256, 256, 0, stream>>>(inputs, h);

  for (int i = 0; i < NBn; ++i) {
    ln_kernel<true><<<Mn / 8, 256, 0, stream>>>(h, ln_g + (2 * i) * Dn,
                                                ln_b + (2 * i) * Dn, lnf, lnb);
    qkv_kernel<<<Mn / 32, 256, 0, stream>>>(lnb,
                                            wbf[0] + i * Dn * Dn,
                                            wbf[1] + i * Dn * Dn,
                                            wbf[2] + i * Dn * Dn,
                                            bia[0] + i * Dn, bia[1] + i * Dn,
                                            bia[2] + i * Dn, qb, kb, vtb);
    attn_kernel<<<(Bn * Hn * Tn / 16) / 8, 256, 0, stream>>>(qb, kb, vtb, ctx);
    gemm128_kernel<0><<<Mn / 32, 256, 0, stream>>>(ctx, wbf[3] + i * Dn * Dn,
                                                   bia[3] + i * Dn, h, nullptr);
    ln_kernel<true><<<Mn / 8, 256, 0, stream>>>(h, ln_g + (2 * i + 1) * Dn,
                                                ln_b + (2 * i + 1) * Dn, lnf, lnb);
    gemm128_kernel<1><<<Mn / 32, 256, 0, stream>>>(lnb, wbf[4] + i * Dn * Dn,
                                                   bia[4] + i * Dn, ffn1, nullptr);
    gemm128_kernel<3><<<Mn / 32, 256, 0, stream>>>(ffn1, wbf[5] + i * Dn * Dn,
                                                   bia[5] + i * Dn, h, lnf);
  }
  // final norm reuses norm_layers[-1] (source quirk)
  ln_kernel<false><<<Mn / 8, 256, 0, stream>>>(h, ln_g + (2 * NBn - 1) * Dn,
                                               ln_b + (2 * NBn - 1) * Dn,
                                               (float*)d_out, nullptr);
}
